// MultiheadLinearAttention_65919158059115
// MI455X (gfx1250) — matrix operand order repaired and checked by reading, not compiled, not run on hardware
//
#include <hip/hip_runtime.h>
#include <cstdint>

// ---------------------------------------------------------------------------
// MultiheadLinearAttention for MI455X (gfx1250, wave32, WMMA bf16 + TDM)
//   B=4, T=4096, D=512, H=8, hd=64, L=1024
// Pipeline:
//   0) fp32 -> bf16 packs (x, Wq, Wc, Wkv) + y transposed to (B,D,T)
//   1) q    = Xb @ Wq^T          (TDM-staged LDS WMMA GEMM)
//   2) kvin = Wc @ Yb            (TDM-staged LDS WMMA GEMM)
//   3) kv   = kvin @ Wkv^T       (direct-global WMMA; k->(B,L,512), v->(B,512,L))
//   4) flash attention per (b,h,16 rows): online softmax, WMMA for q.k^T and P.V
// ---------------------------------------------------------------------------

typedef __attribute__((ext_vector_type(16))) __bf16 v16bf;
typedef __attribute__((ext_vector_type(8)))  __bf16 v8bf;
typedef __attribute__((ext_vector_type(8)))  float  v8f;
typedef __attribute__((ext_vector_type(4)))  unsigned int v4u;
typedef __attribute__((ext_vector_type(8)))  int v8i;

#define TSEQ   4096
#define BATCH  4
#define DMODEL 512
#define NHEADS 8
#define HD     64
#define LCOMP  1024
#define KS     64      // K-panel staged per TDM load

// ---------------- fragment loaders (ISA 7.12.2 layouts) --------------------
// A 16x32 bf16: lanes 0-15 row M=lane, K = k+0..7 & k+16..23
//               lanes 16-31 row M=lane-16, K = k+8..15 & k+24..31
__device__ __forceinline__ v16bf load_afrag(const __bf16* rowbase, int lane, int k) {
  const __bf16* p = rowbase + k + ((lane >> 4) << 3);
  v8bf lo = *(const v8bf*)(p);
  v8bf hi = *(const v8bf*)(p + 16);
  v16bf r;
#pragma unroll
  for (int i = 0; i < 8; ++i) { r[i] = lo[i]; r[i + 8] = hi[i]; }
  return r;
}

// B 32x16 bf16: lane holds column n; lanes 0-15 K=k+0..7 & k+16..23, lanes 16-31 K=k+8..15 & k+24..31
__device__ __forceinline__ v16bf load_bfrag(const __bf16* rowbase, int lane, int k) {
  const __bf16* p = rowbase + k + ((lane >> 4) << 3);
  v8bf lo = *(const v8bf*)(p);
  v8bf hi = *(const v8bf*)(p + 16);
  v16bf r;
#pragma unroll
  for (int i = 0; i < 8; ++i) { r[i] = lo[i]; r[i + 8] = hi[i]; }
  return r;
}

// ---------------- TDM: 2D tile load Global -> LDS (ISA ch.8 D#) ------------
// rows x cols tile of 2-byte elements, row stride `stride_elems`, tile start
// at `gaddr`, deposited row-major contiguous at LDS byte offset `lds_off`.
__device__ __forceinline__ void tdm_load_tile_2d(uint32_t lds_off, const void* gaddr,
                                                 uint32_t rows, uint32_t cols,
                                                 uint64_t stride_elems) {
  uint64_t ga = (uint64_t)(size_t)gaddr;
  v4u g0;
  g0[0] = 1u;                                    // count=1 valid descriptor
  g0[1] = lds_off;                               // lds_addr (bytes)
  g0[2] = (uint32_t)ga;                          // global_addr[31:0]
  g0[3] = (uint32_t)((ga >> 32) & 0x1FFFFFFu)    // global_addr[56:32]
          | (2u << 30);                          // type = 2 ("image")
  v8i g1;
  g1[0] = (int)(1u << 16);                       // wg_mask=0, data_size=1 (2B)
  g1[1] = (int)((cols & 0xFFFFu) << 16);         // tensor_dim0[15:0]
  g1[2] = (int)(((cols >> 16) & 0xFFFFu)         // tensor_dim0[31:16]
                | ((rows & 0xFFFFu) << 16));     // tensor_dim1[15:0]
  g1[3] = (int)(((rows >> 16) & 0xFFFFu)         // tensor_dim1[31:16]
                | ((cols & 0xFFFFu) << 16));     // tile_dim0
  g1[4] = (int)(rows & 0xFFFFu);                 // tile_dim1 (tile_dim2 = 0)
  g1[5] = (int)(stride_elems & 0xFFFFFFFFu);     // tensor_dim0_stride[31:0]
  g1[6] = (int)((stride_elems >> 32) & 0xFFFFu); // tensor_dim0_stride[47:32]
  g1[7] = 0;                                     // tensor_dim1_stride hi (unused, 2D)
  asm volatile("tensor_load_to_lds %0, %1" :: "s"(g0), "s"(g1) : "memory");
}

// ---------------- pack kernels --------------------------------------------
__global__ void cvt_f32_bf16(const float* __restrict__ src, __bf16* __restrict__ dst, int n) {
  int i = blockIdx.x * blockDim.x + threadIdx.x;
  if (i < n) dst[i] = (__bf16)src[i];
}

// y: (T,B,D) fp32 -> ybt: (B,D,T) bf16
__global__ void transpose_y(const float* __restrict__ y, __bf16* __restrict__ ybt, int n) {
  int idx = blockIdx.x * blockDim.x + threadIdx.x;
  if (idx >= n) return;
  int d = idx % DMODEL;
  int r = idx / DMODEL;
  int b = r % BATCH;
  int t = r / BATCH;
  ybt[((size_t)b * DMODEL + d) * TSEQ + t] = (__bf16)y[idx];
}

// ---------------- TDM-staged cooperative WMMA GEMM:  C = A * B^T -----------
// 4 waves / workgroup, 64x64 C tile; A,B K-panels staged in LDS by the TDM.
// A rows: M x K (lda). B rows: N x K (ldb). K multiple of 64.
__global__ __launch_bounds__(128)
void gemm_bf16_tdm(const __bf16* __restrict__ A, long lda, long bsA,
                   const __bf16* __restrict__ Bm, long ldb, long bsB,
                   __bf16* __restrict__ C, long ldc, long bsC, int K) {
  __shared__ __align__(16) __bf16 Abuf[64 * KS];
  __shared__ __align__(16) __bf16 Bbuf[64 * KS];

  const int lane = threadIdx.x & 31;
  const int wave = threadIdx.x >> 5;
  const int m0 = blockIdx.y << 6;
  const int n0 = blockIdx.x << 6;
  const int b  = blockIdx.z;

  const __bf16* Ab = A + (size_t)b * bsA;
  const __bf16* Bb = Bm + (size_t)b * bsB;
  const uint32_t lds_a = (uint32_t)(size_t)(void*)Abuf;
  const uint32_t lds_b = (uint32_t)(size_t)(void*)Bbuf;

  v8f acc[4] = {};

  for (int k = 0; k < K; k += KS) {
    if (wave == 0) {
      tdm_load_tile_2d(lds_a, Ab + (size_t)m0 * lda + k, 64, KS, (uint64_t)lda);
      tdm_load_tile_2d(lds_b, Bb + (size_t)n0 * ldb + k, 64, KS, (uint64_t)ldb);
      __builtin_amdgcn_s_wait_tensorcnt(0);
    }
    __syncthreads();

    const __bf16* Arow = Abuf + (size_t)((wave << 4) + (lane & 15)) * KS;
#pragma unroll
    for (int kk = 0; kk < KS; kk += 32) {
      v16bf a = load_afrag(Arow, lane, kk);
#pragma unroll
      for (int j = 0; j < 4; ++j) {
        const __bf16* Brow = Bbuf + (size_t)((j << 4) + (lane & 15)) * KS;
        v16bf bf = load_bfrag(Brow, lane, kk);
        acc[j] = __builtin_amdgcn_wmma_f32_16x16x32_bf16(
            false, a, false, bf, (short)0, acc[j], false, false);
      }
    }
    __syncthreads();
  }

  __bf16* Cb = C + (size_t)b * bsC;
  const int rbase = m0 + (wave << 4) + ((lane >> 4) << 3);
  const int c = lane & 15;
#pragma unroll
  for (int i = 0; i < 8; ++i)
#pragma unroll
    for (int j = 0; j < 4; ++j)
      Cb[(size_t)(rbase + i) * ldc + n0 + j * 16 + c] = (__bf16)acc[j][i];
}

// ---------------- GEMM3 with split k / v-transposed store ------------------
// kvin: (B,L,512), Wkv: (1024,512).  n<512 -> karr (B,L,512); n>=512 -> vT (B,512,L)
__global__ __launch_bounds__(32)
void gemm_kv(const __bf16* __restrict__ kvin, const __bf16* __restrict__ wkv,
             __bf16* __restrict__ karr, __bf16* __restrict__ vT) {
  const int lane = threadIdx.x;
  const int m0 = blockIdx.y << 4;   // l
  const int n0 = blockIdx.x << 6;   // output channel
  const int b  = blockIdx.z;

  const __bf16* Arow = kvin + ((size_t)b * LCOMP + m0 + (lane & 15)) * DMODEL;
  const __bf16* Brow[4];
#pragma unroll
  for (int j = 0; j < 4; ++j)
    Brow[j] = wkv + (size_t)(n0 + j * 16 + (lane & 15)) * DMODEL;

  v8f acc[4] = {};
  for (int k = 0; k < DMODEL; k += 32) {
    v16bf a = load_afrag(Arow, lane, k);
#pragma unroll
    for (int j = 0; j < 4; ++j) {
      v16bf bf = load_bfrag(Brow[j], lane, k);
      acc[j] = __builtin_amdgcn_wmma_f32_16x16x32_bf16(
          false, a, false, bf, (short)0, acc[j], false, false);
    }
  }

  const int rbase = m0 + ((lane >> 4) << 3);
  const int c = lane & 15;
#pragma unroll
  for (int i = 0; i < 8; ++i)
#pragma unroll
    for (int j = 0; j < 4; ++j) {
      int n = n0 + j * 16 + c;
      __bf16 v = (__bf16)acc[j][i];
      if (n < DMODEL)
        karr[((size_t)b * LCOMP + rbase + i) * DMODEL + n] = v;
      else
        vT[((size_t)b * DMODEL + (n - DMODEL)) * LCOMP + rbase + i] = v;
    }
}

// ---------------- flash attention over compressed keys ---------------------
// one wave per (b, h, 16 q-rows); L processed in chunks of 32
__global__ __launch_bounds__(32)
void attn_kernel(const __bf16* __restrict__ qbf, const __bf16* __restrict__ karr,
                 const __bf16* __restrict__ vT, float* __restrict__ out) {
  __shared__ __align__(16) __bf16 p_lds[16][32];

  const int lane = threadIdx.x;
  const int t0 = blockIdx.x << 4;
  const int h  = blockIdx.y;
  const int b  = blockIdx.z;
  const float scale = 0.125f;  // 1/sqrt(64)

  // q tile (16 x 64) as two A fragments
  const __bf16* qrow = qbf + ((size_t)b * TSEQ + t0 + (lane & 15)) * DMODEL + h * HD;
  v16bf qa0 = load_afrag(qrow, lane, 0);
  v16bf qa1 = load_afrag(qrow, lane, 32);

  float mrow[8], srow[8];
  v8f o[4] = {};
#pragma unroll
  for (int i = 0; i < 8; ++i) { mrow[i] = -1e30f; srow[i] = 0.0f; }

  for (int l0 = 0; l0 < LCOMP; l0 += 32) {
    // scores: two 16x16 tiles (K = 64)
    v8f s[2] = {};
#pragma unroll
    for (int jt = 0; jt < 2; ++jt) {
      const __bf16* krow =
          karr + ((size_t)b * LCOMP + l0 + jt * 16 + (lane & 15)) * DMODEL + h * HD;
      v16bf kb0 = load_bfrag(krow, lane, 0);
      v16bf kb1 = load_bfrag(krow, lane, 32);
      s[jt] = __builtin_amdgcn_wmma_f32_16x16x32_bf16(
          false, qa0, false, kb0, (short)0, s[jt], false, false);
      s[jt] = __builtin_amdgcn_wmma_f32_16x16x32_bf16(
          false, qa1, false, kb1, (short)0, s[jt], false, false);
    }

    // online softmax update (rows: VGPR i -> M = i + 8*(lane>>4); cols in lane&15)
#pragma unroll
    for (int i = 0; i < 8; ++i) {
      float s0 = s[0][i] * scale;
      float s1 = s[1][i] * scale;
      float cm = fmaxf(s0, s1);
#pragma unroll
      for (int d = 1; d < 16; d <<= 1)
        cm = fmaxf(cm, __shfl_xor(cm, d, 32));
      float mnew  = fmaxf(mrow[i], cm);
      float alpha = __expf(mrow[i] - mnew);
      float p0 = __expf(s0 - mnew);
      float p1 = __expf(s1 - mnew);
      float cs = p0 + p1;
#pragma unroll
      for (int d = 1; d < 16; d <<= 1)
        cs += __shfl_xor(cs, d, 32);
      srow[i] = srow[i] * alpha + cs;
      mrow[i] = mnew;
#pragma unroll
      for (int j = 0; j < 4; ++j) o[j][i] *= alpha;

      const int r = i + ((lane >> 4) << 3);
      const int c = lane & 15;
      p_lds[r][c]      = (__bf16)p0;
      p_lds[r][c + 16] = (__bf16)p1;
    }
    asm volatile("s_wait_dscnt 0" ::: "memory");

    // reload P as A fragment (16x32) from LDS per ISA A-layout
    {
      const int m  = lane & 15;
      const int kb = (lane >> 4) << 3;
      v16bf pa;
#pragma unroll
      for (int i = 0; i < 8; ++i) {
        pa[i]     = p_lds[m][kb + i];
        pa[i + 8] = p_lds[m][kb + 16 + i];
      }
      // o += P @ Vchunk  (V rows are contiguous in l thanks to vT layout)
#pragma unroll
      for (int j = 0; j < 4; ++j) {
        const __bf16* vrow =
            vT + ((size_t)b * DMODEL + h * HD + j * 16 + (lane & 15)) * LCOMP + l0;
        v16bf vb = load_bfrag(vrow, lane, 0);
        o[j] = __builtin_amdgcn_wmma_f32_16x16x32_bf16(
            false, pa, false, vb, (short)0, o[j], false, false);
      }
    }
  }

  // normalize and store: out layout (T,B,D) fp32
#pragma unroll
  for (int i = 0; i < 8; ++i) {
    float inv = 1.0f / srow[i];
    int t = t0 + i + ((lane >> 4) << 3);
    int c = lane & 15;
#pragma unroll
    for (int j = 0; j < 4; ++j)
      out[((size_t)t * BATCH + b) * DMODEL + h * HD + j * 16 + c] = o[j][i] * inv;
  }
}

// ---------------------------------------------------------------------------
extern "C" void kernel_launch(void* const* d_in, const int* in_sizes, int n_in,
                              void* d_out, int out_size, void* d_ws, size_t ws_size,
                              hipStream_t stream) {
  const float* x   = (const float*)d_in[0];
  const float* y   = (const float*)d_in[1];
  // d_in[2] = z, unused by the reference
  const float* Wq  = (const float*)d_in[3];
  const float* Wc  = (const float*)d_in[4];
  const float* Wkv = (const float*)d_in[5];
  float* out = (float*)d_out;

  // workspace carve-up (bf16, all region sizes multiples of 16 elements)
  size_t off = 0;
  auto carve = [&](size_t nelem) {
    __bf16* p = (__bf16*)((char*)d_ws + off);
    off += nelem * sizeof(__bf16);
    return p;
  };
  __bf16* xbf  = carve((size_t)TSEQ * BATCH * DMODEL);   // (T,B,D)
  __bf16* ybt  = carve((size_t)BATCH * DMODEL * TSEQ);   // (B,D,T)
  __bf16* wq   = carve((size_t)DMODEL * DMODEL);         // (D,D)
  __bf16* wc   = carve((size_t)LCOMP * TSEQ);            // (L,T)
  __bf16* wkv  = carve((size_t)2 * DMODEL * DMODEL);     // (2D,D)
  __bf16* qb   = carve((size_t)BATCH * TSEQ * DMODEL);   // (B,T,D)
  __bf16* kvin = carve((size_t)BATCH * LCOMP * DMODEL);  // (B,L,D)
  __bf16* karr = carve((size_t)BATCH * LCOMP * DMODEL);  // (B,L,D)
  __bf16* vt   = carve((size_t)BATCH * DMODEL * LCOMP);  // (B,D,L)

  int n;
  n = TSEQ * BATCH * DMODEL;
  cvt_f32_bf16<<<(n + 255) / 256, 256, 0, stream>>>(x, xbf, n);
  transpose_y<<<(n + 255) / 256, 256, 0, stream>>>(y, ybt, n);
  n = DMODEL * DMODEL;
  cvt_f32_bf16<<<(n + 255) / 256, 256, 0, stream>>>(Wq, wq, n);
  n = LCOMP * TSEQ;
  cvt_f32_bf16<<<(n + 255) / 256, 256, 0, stream>>>(Wc, wc, n);
  n = 2 * DMODEL * DMODEL;
  cvt_f32_bf16<<<(n + 255) / 256, 256, 0, stream>>>(Wkv, wkv, n);

  // q = Xb @ Wq^T   (M=T, N=D, K=D) per batch  [TDM-staged]
  gemm_bf16_tdm<<<dim3(DMODEL / 64, TSEQ / 64, BATCH), 128, 0, stream>>>(
      xbf, (long)BATCH * DMODEL, (long)DMODEL,
      wq, DMODEL, 0,
      qb, DMODEL, (long)TSEQ * DMODEL, DMODEL);

  // kvin = Wc @ Yb  (M=L, N=D, K=T) per batch  [TDM-staged]
  gemm_bf16_tdm<<<dim3(DMODEL / 64, LCOMP / 64, BATCH), 128, 0, stream>>>(
      wc, TSEQ, 0,
      ybt, TSEQ, (long)DMODEL * TSEQ,
      kvin, DMODEL, (long)LCOMP * DMODEL, TSEQ);

  // kv = kvin @ Wkv^T, split into k and v^T
  gemm_kv<<<dim3(2 * DMODEL / 64, LCOMP / 16, BATCH), 32, 0, stream>>>(
      kvin, wkv, karr, vt);

  // attention
  attn_kernel<<<dim3(TSEQ / 16, NHEADS, BATCH), 32, 0, stream>>>(qb, karr, vt, out);
}
